// KeyedGRU_30537217475010
// MI455X (gfx1250) — matrix operand order repaired and checked by reading, not compiled, not run on hardware
//
#include <hip/hip_runtime.h>
#include <stdint.h>

typedef _Float16 v16h __attribute__((ext_vector_type(16)));
typedef _Float16 v8h  __attribute__((ext_vector_type(8)));
typedef float    v8f  __attribute__((ext_vector_type(8)));
typedef unsigned int u32x4 __attribute__((ext_vector_type(4)));
typedef int      i32x8 __attribute__((ext_vector_type(8)));
typedef int      i32x4 __attribute__((ext_vector_type(4)));

#define B_  64
#define T_  2048
#define I_  256
#define H_  256
#define G3  768      // 3*H
#define KL_ 16
#define KB_ 4
#define HROW 264     // padded LDS row stride (halves) for h: 528B -> conflict-free b128 loads

// workspace layout (bytes)
#define GI_BYTES   ((size_t)T_ * B_ * G3 * 2)            // 201,326,592
#define WIH_OFF    (GI_BYTES)
#define WHH_OFF    (WIH_OFF + (size_t)G3 * I_ * 2)
#define GATES_OFF  (WHH_OFF + (size_t)G3 * H_ * 2)

// ---------------- weight f32 -> f16 conversion ----------------
__global__ void k_cvt(const float* __restrict__ src, _Float16* __restrict__ dst, int n) {
  int i = blockIdx.x * blockDim.x + threadIdx.x;
  if (i < n) dst[i] = (_Float16)src[i];
}

__device__ __forceinline__ float sigmoidf_(float x) {
  return 1.0f / (1.0f + __expf(-x));
}

// ---------------- key-gate scan: 16 tiny GRU steps, KB=4 ----------------
__global__ __launch_bounds__(256) void k_keygate(
    const float* __restrict__ wm_key, const float* __restrict__ wih,
    const float* __restrict__ whh, const float* __restrict__ bih,
    const float* __restrict__ bhh, float* __restrict__ gates) {
  __shared__ float h[KB_][H_];
  const int j = threadIdx.x;
  for (int kb = 0; kb < KB_; ++kb) h[kb][j] = 0.0f;
  __syncthreads();
  const float bi0 = bih[j], bi1 = bih[H_ + j], bi2 = bih[2 * H_ + j];
  const float bh0 = bhh[j], bh1 = bhh[H_ + j], bh2 = bhh[2 * H_ + j];
  for (int l = 0; l < KL_; ++l) {
    float ai[3][KB_] = {}, ah[3][KB_] = {};
    for (int g = 0; g < 3; ++g) {
      const float* wi = wih + (size_t)(g * H_ + j) * I_;
      const float* wh = whh + (size_t)(g * H_ + j) * H_;
      for (int i = 0; i < I_; ++i) {
        float wiv = wi[i], whv = wh[i];
        #pragma unroll
        for (int kb = 0; kb < KB_; ++kb) {
          ai[g][kb] = fmaf(wiv, wm_key[((size_t)kb * KL_ + l) * I_ + i], ai[g][kb]);
          ah[g][kb] = fmaf(whv, h[kb][i], ah[g][kb]);
        }
      }
    }
    float hy[KB_], rsum = 0.0f;
    #pragma unroll
    for (int kb = 0; kb < KB_; ++kb) {
      float r  = sigmoidf_(ai[0][kb] + bi0 + ah[0][kb] + bh0);
      float ig = sigmoidf_(ai[1][kb] + bi1 + ah[1][kb] + bh1);
      float n  = tanhf(ai[2][kb] + bi2 + r * (ah[2][kb] + bh2));
      hy[kb] = n + ig * (h[kb][j] - n);
      rsum += r;
    }
    __syncthreads();
    for (int kb = 0; kb < KB_; ++kb) h[kb][j] = hy[kb];
    gates[l * H_ + j] = rsum * 0.25f;
    __syncthreads();
  }
}

// ---------------- gi = x @ W_ih^T + b_ih  (full-chip WMMA GEMM) ----------------
// gi layout: [t][b][768] f16.  Block = one 16-row M-tile (same t), 8 waves x 6 N-tiles.
__global__ __launch_bounds__(256) void k_gi_gemm(
    const float* __restrict__ x, const _Float16* __restrict__ wih16,
    const float* __restrict__ bih, _Float16* __restrict__ gi) {
  const int mt = blockIdx.x;           // 0..8191
  const int wv = threadIdx.x >> 5;     // 0..7
  const int ln = threadIdx.x & 31;
  const int hi = ln >> 4;              // lane half
  const int nc = ln & 15;
  const int t  = mt >> 2;
  const int b0 = (mt & 3) * 16;
  const float* xrow = x + ((size_t)(b0 + nc) * T_ + t) * I_;  // x[b][t][:]
  v8f acc[6] = {};
  #pragma unroll
  for (int k = 0; k < 8; ++k) {
    union { v16h v; _Float16 e[16]; } A;
    const int i0 = k * 32 + hi * 16;
    #pragma unroll
    for (int e = 0; e < 8; ++e) A.e[e] = (_Float16)xrow[i0 + e];
    #pragma unroll
    for (int e = 0; e < 8; ++e) A.e[8 + e] = (_Float16)xrow[i0 + 8 + e];
    #pragma unroll
    for (int q = 0; q < 6; ++q) {
      const int ncol = (wv * 6 + q) * 16 + nc;
      v16h Bf = *(const v16h*)(wih16 + (size_t)ncol * I_ + k * 32 + hi * 16);
      acc[q] = __builtin_amdgcn_wmma_f32_16x16x32_f16(false, A.v, false, Bf,
                                                      (short)0, acc[q], false, false);
    }
  }
  const int rbase = mt * 16 + hi * 8;
  #pragma unroll
  for (int q = 0; q < 6; ++q) {
    const int ncol = (wv * 6 + q) * 16 + nc;
    const float bias = bih[ncol];
    #pragma unroll
    for (int v = 0; v < 8; ++v)
      gi[(size_t)(rbase + v) * G3 + ncol] = (_Float16)(acc[q][v] + bias);
  }
}

// ---------------- TDM: load one [64][768] f16 gi tile global -> LDS ----------------
__device__ __forceinline__ void tdm_gi(const _Float16* gsrc, uint32_t ldsoff) {
  uint64_t ga = (uint64_t)(uintptr_t)gsrc;
  // D# group0: count=1 | lds_addr | global_addr | type=2
  u32x4 g0 = { 1u, ldsoff, (uint32_t)ga, (uint32_t)(ga >> 32) | 0x80000000u };
  // D# group1: data_size=2B; tensor_dim0=768, tensor_dim1=64; tile 768x64; stride0=768
  i32x8 g1 = { (int)0x00010000u, (int)(768u << 16), (int)(64u << 16),
               (int)(768u << 16), 64, 768, 0, 0 };
  i32x4 z4 = { 0, 0, 0, 0 };
  i32x8 z8 = { 0, 0, 0, 0, 0, 0, 0, 0 };
  __builtin_amdgcn_tensor_load_to_lds(g0, g1, z4, z4, z8, 0);
}

// ---------------- persistent single-WGP GRU scan ----------------
// 512 threads = 16 waves; wave w owns H-column tile jt=w, all 4 batch tiles,
// with its 24 W_hh^T B-fragments (r/i/n x 8 k-blocks) resident in VGPRs.
__global__ __launch_bounds__(512) void k_scan(
    const _Float16* __restrict__ gi_g, const _Float16* __restrict__ whh16,
    const float* __restrict__ bhh, const float* __restrict__ gates,
    float* __restrict__ out) {
  extern __shared__ char smem[];
  _Float16* sh_h  = (_Float16*)smem;                              // 2 x [64][HROW]
  _Float16* sh_gi = (_Float16*)(smem + 2 * 64 * HROW * 2);        // 2 x [64][768]
  const uint32_t gi_lds_off = (uint32_t)(uintptr_t)smem + 2u * 64u * HROW * 2u;

  const int wv = threadIdx.x >> 5;   // = jt (0..15)
  const int ln = threadIdx.x & 31;
  const int hi = ln >> 4;
  const int nc = ln & 15;
  const int j  = wv * 16 + nc;       // hidden column owned by this lane

  for (int idx = threadIdx.x; idx < 2 * 64 * HROW; idx += 512) sh_h[idx] = (_Float16)0.0f;

  const float bR = bhh[j], bI = bhh[H_ + j], bN = bhh[2 * H_ + j];
  float gk[KL_];
  #pragma unroll
  for (int l = 0; l < KL_; ++l) gk[l] = gates[l * H_ + j];

  // resident W_hh^T B-fragments: B[k][n] = W_hh[gatecol][k]; lane's col = g*256+j
  v16h Br[8], Bi[8], Bn[8];
  #pragma unroll
  for (int k = 0; k < 8; ++k) {
    const int krun = k * 32 + hi * 16;
    Br[k] = *(const v16h*)(whh16 + (size_t)(0 * H_ + j) * H_ + krun);
    Bi[k] = *(const v16h*)(whh16 + (size_t)(1 * H_ + j) * H_ + krun);
    Bn[k] = *(const v16h*)(whh16 + (size_t)(2 * H_ + j) * H_ + krun);
  }

  if (wv == 0) {  // prefetch gi tile for t=0
    tdm_gi(gi_g, gi_lds_off);
    __builtin_amdgcn_s_wait_tensorcnt(0);
  }
  __syncthreads();

  for (int t = 0; t < T_; ++t) {
    const int pb = t & 1, nb = pb ^ 1;
    if (wv == 0 && t + 1 < T_)  // async prefetch of next gi tile, overlapped with WMMAs
      tdm_gi(gi_g + (size_t)(t + 1) * B_ * G3, gi_lds_off + (uint32_t)nb * B_ * G3 * 2);

    const _Float16* hin = sh_h + pb * 64 * HROW;
    _Float16* hout      = sh_h + nb * 64 * HROW;

    v8f aR[4] = {}, aI[4] = {}, aN[4] = {};
    #pragma unroll
    for (int k = 0; k < 8; ++k) {
      union { v16h v; v8h h2[2]; } A[4];
      #pragma unroll
      for (int m = 0; m < 4; ++m) {
        const _Float16* ap = hin + (m * 16 + nc) * HROW + k * 32 + hi * 16;
        A[m].h2[0] = *(const v8h*)ap;
        A[m].h2[1] = *(const v8h*)(ap + 8);
      }
      #pragma unroll
      for (int m = 0; m < 4; ++m) {
        aR[m] = __builtin_amdgcn_wmma_f32_16x16x32_f16(false, A[m].v, false, Br[k], (short)0, aR[m], false, false);
        aI[m] = __builtin_amdgcn_wmma_f32_16x16x32_f16(false, A[m].v, false, Bi[k], (short)0, aI[m], false, false);
        aN[m] = __builtin_amdgcn_wmma_f32_16x16x32_f16(false, A[m].v, false, Bn[k], (short)0, aN[m], false, false);
      }
    }

    const _Float16* gi_l = sh_gi + pb * B_ * G3;
    const float g = (t < KL_) ? gk[t] : 1.0f;
    #pragma unroll
    for (int m = 0; m < 4; ++m) {
      #pragma unroll
      for (int v = 0; v < 8; ++v) {
        const int b = m * 16 + v + hi * 8;          // C layout: M = v + 8*(lane>=16)
        const float giR = (float)gi_l[b * G3 + j];
        const float giI = (float)gi_l[b * G3 + H_ + j];
        const float giN = (float)gi_l[b * G3 + 2 * H_ + j];
        const float hprev = (float)hin[b * HROW + j];
        const float r  = sigmoidf_(giR + aR[m][v] + bR);
        const float ig = sigmoidf_(giI + aI[m][v] + bI);
        const float n  = tanhf(giN + r * (aN[m][v] + bN));
        const float hy = n + ig * (hprev - n);
        out[((size_t)t * B_ + b) * H_ + j] = hy;     // pre-gate output
        hout[b * HROW + j] = (_Float16)(hy * g);     // gated carry
      }
    }

    if (wv == 0) __builtin_amdgcn_s_wait_tensorcnt(0);
    __syncthreads();
  }
}

extern "C" void kernel_launch(void* const* d_in, const int* in_sizes, int n_in,
                              void* d_out, int out_size, void* d_ws, size_t ws_size,
                              hipStream_t stream) {
  const float* x      = (const float*)d_in[0];
  const float* wm_key = (const float*)d_in[1];
  const float* wih    = (const float*)d_in[2];
  const float* whh    = (const float*)d_in[3];
  const float* bih    = (const float*)d_in[4];
  const float* bhh    = (const float*)d_in[5];
  float* out = (float*)d_out;

  char* ws = (char*)d_ws;
  _Float16* gi    = (_Float16*)ws;
  _Float16* wih16 = (_Float16*)(ws + WIH_OFF);
  _Float16* whh16 = (_Float16*)(ws + WHH_OFF);
  float*    gatesb = (float*)(ws + GATES_OFF);

  k_cvt<<<(G3 * I_ + 255) / 256, 256, 0, stream>>>(wih, wih16, G3 * I_);
  k_cvt<<<(G3 * H_ + 255) / 256, 256, 0, stream>>>(whh, whh16, G3 * H_);
  k_keygate<<<1, 256, 0, stream>>>(wm_key, wih, whh, bih, bhh, gatesb);
  k_gi_gemm<<<(T_ * B_) / 16, 256, 0, stream>>>(x, wih16, bih, gi);

  const size_t smem_bytes = 2 * 64 * HROW * 2 + 2 * (size_t)B_ * G3 * 2;  // 264,192 B
  k_scan<<<1, 512, smem_bytes, stream>>>(gi, whh16, bhh, gatesb, out);
}
